// ReactivityGAT_54417235640493
// MI455X (gfx1250) — matrix operand order repaired and checked by reading, not compiled, not run on hardware
//
#include <hip/hip_runtime.h>
#include <hip/hip_bf16.h>

// ---------------------------------------------------------------------------
// ReactivityGAT forward for MI455X (gfx1250, wave32).
// Dense GEMMs -> v_wmma_f32_16x16x32_f16 (f16 in / f32 accumulate), operands
// pre-converted to padded f16 so the WMMA inner loop is pure b128 loads + wmma
// (no per-element guards / exec-mask churn).
// Sparse parts -> native f32 global atomics + order-preserving uint max keys.
// d_in mapping assumes jax pytree flatten of params (dict keys sorted):
//   0:x 1:edge_index 2:edge_attr 3:batch 4:global_features
//   5..12: fc0.W,fc0.b,fc1.W,fc1.b,fc2.W,fc2.b,fc3.W,fc3.b
//   13..22: L0 {W,We,att_dst,att_edge,att_src,bias,ln_b,ln_g,proj_W,proj_b}
//   23..30: L1 {W,We,att_dst,att_edge,att_src,bias,ln_b,ln_g}
//   31..38: L2 {same}
//   39: node_W  40: node_b
// ---------------------------------------------------------------------------

#define NNODES   20000
#define NEDGES   320000
#define NEA      (NNODES + NEDGES)   // edges + self loops
#define NGRAPH   128
#define HID      64
#define HEADS    4
#define WIDTH    256
#define NODE_DIM 58
#define EDGE_DIM 13
#define GDIM     526                 // 256 + 256 + 14
#define NEG_SLOPE 0.2f

typedef __attribute__((ext_vector_type(16))) _Float16 v16h;
typedef __attribute__((ext_vector_type(8)))  _Float16 v8h;
typedef __attribute__((ext_vector_type(8)))  float    v8f;

static inline int padup(int x, int a) { return (x + a - 1) & ~(a - 1); }
static inline int cdiv(int a, int b) { return (a + b - 1) / b; }

// ----- order-preserving float<->uint key for atomic max -----
__device__ __forceinline__ unsigned fkey(float f) {
  unsigned b = __float_as_uint(f);
  return (b & 0x80000000u) ? ~b : (b | 0x80000000u);
}
__device__ __forceinline__ float fkey_inv(unsigned k) {
  return (k & 0x80000000u) ? __uint_as_float(k & 0x7FFFFFFFu)
                           : __uint_as_float(~k);
}

// ---------------------------------------------------------------------------
// Operand conversion: f32 -> padded f16.
//   cvtA : A[M,K] f32 -> Ah[M,Kp] f16 (K zero-padded to Kp, mult of 32)
//   cvtBt: B[K,N] f32 -> Bt[Np,Kp] f16 (transposed; zero-padded both dims)
// ---------------------------------------------------------------------------
__global__ void cvtA(const float* __restrict__ A, _Float16* __restrict__ Ah,
                     int M, int K, int Kp) {
  int i = blockIdx.x * blockDim.x + threadIdx.x;
  if (i >= M * Kp) return;
  const int row = i / Kp, k = i - row * Kp;
  Ah[i] = (k < K) ? (_Float16)A[(size_t)row * K + k] : (_Float16)0.f;
}
__global__ void cvtBt(const float* __restrict__ B, _Float16* __restrict__ Bt,
                      int K, int N, int Kp, int Np) {
  int i = blockIdx.x * blockDim.x + threadIdx.x;
  if (i >= Np * Kp) return;
  const int col = i / Kp, k = i - col * Kp;
  Bt[i] = (k < K && col < N) ? (_Float16)B[(size_t)k * N + col] : (_Float16)0.f;
}

// ---------------------------------------------------------------------------
// WMMA GEMM on pre-converted operands: C[M,N] = Ah[M,Kp] @ Bt[Np,Kp]^T.
// One wave computes a 16x32 C tile (two 16x16 WMMA accumulators sharing A).
// Requirements: M % 16 == 0, Kp % 32 == 0, Np % 32 == 0 (pads are zero).
// Fragment layouts used here (wave32; A and B share one contiguous K order):
//   A 16x32 f16 : row=lane&15, elem  j   -> K=half*16+j
//   B 32x16 f16 : col=lane&15, elem  j   -> K=half*16+j
//   C 16x16 f32 : col=lane&15, elem  i   -> row=half*8+i
// ---------------------------------------------------------------------------
__global__ __launch_bounds__(32)
void wmma_gemm_f16(const _Float16* __restrict__ A, const _Float16* __restrict__ Bt,
                   const float* __restrict__ bias, float* __restrict__ C,
                   int M, int N, int Kp, int relu)
{
  const int lane = threadIdx.x & 31;
  const int half = lane >> 4;
  const int l15  = lane & 15;
  const int row0 = blockIdx.x << 4;
  const int col0 = blockIdx.y << 5;   // 32 columns per wave

  const _Float16* Ap  = A  + (size_t)(row0 + l15) * Kp + half * 16;
  const _Float16* Bp0 = Bt + (size_t)(col0 + l15) * Kp + half * 16;
  const _Float16* Bp1 = Bp0 + (size_t)16 * Kp;

  v8f acc0 = {}, acc1 = {};
  for (int k0 = 0; k0 < Kp; k0 += 32) {
    const v8h a0  = *(const v8h*)(Ap  + k0);        // K = k0 + half*16 + 0..7
    const v8h a1  = *(const v8h*)(Ap  + k0 + 8);    // K = k0 + half*16 + 8..15
    const v8h b0l = *(const v8h*)(Bp0 + k0);        // K = k0 + half*16 + 0..7
    const v8h b0h = *(const v8h*)(Bp0 + k0 + 8);    // K = k0 + half*16 + 8..15
    const v8h b1l = *(const v8h*)(Bp1 + k0);
    const v8h b1h = *(const v8h*)(Bp1 + k0 + 8);
    const v16h a  = __builtin_shufflevector(a0, a1, 0,1,2,3,4,5,6,7,8,9,10,11,12,13,14,15);
    const v16h b0 = __builtin_shufflevector(b0l, b0h, 0,1,2,3,4,5,6,7,8,9,10,11,12,13,14,15);
    const v16h b1 = __builtin_shufflevector(b1l, b1h, 0,1,2,3,4,5,6,7,8,9,10,11,12,13,14,15);
    acc0 = __builtin_amdgcn_wmma_f32_16x16x32_f16(false, a, false, b0, (short)0, acc0, false, false);
    acc1 = __builtin_amdgcn_wmma_f32_16x16x32_f16(false, a, false, b1, (short)0, acc1, false, false);
  }

  const int c0 = col0 + l15;
  const int c1 = col0 + 16 + l15;
  const float bb0 = (bias && c0 < N) ? bias[c0] : 0.f;
  const float bb1 = (bias && c1 < N) ? bias[c1] : 0.f;
  const int rbase = row0 + half * 8;            // rows always < M (M % 16 == 0)
  if (c0 < N) {
#pragma unroll
    for (int i = 0; i < 8; ++i) {
      float v = acc0[i] + bb0;
      if (relu) v = fmaxf(v, 0.f);
      C[(size_t)(rbase + i) * N + c0] = v;
    }
  }
  if (c1 < N) {
#pragma unroll
    for (int i = 0; i < 8; ++i) {
      float v = acc1[i] + bb1;
      if (relu) v = fmaxf(v, 0.f);
      C[(size_t)(rbase + i) * N + c1] = v;
    }
  }
}

// ---------------------------------------------------------------------------
// Fills
// ---------------------------------------------------------------------------
__global__ void fill_f32(float* p, float v, int n) {
  int i = blockIdx.x * blockDim.x + threadIdx.x;
  if (i < n) p[i] = v;
}
__global__ void fill_u32(unsigned* p, unsigned v, int n) {
  int i = blockIdx.x * blockDim.x + threadIdx.x;
  if (i < n) p[i] = v;
}

// ---------------------------------------------------------------------------
// Self-loop edge attrs: loop_attr[n] = segsum(edge_attr by dst) / max(deg,1)
// ---------------------------------------------------------------------------
__global__ void loop_attr_accum(const int* __restrict__ dst,
                                const float* __restrict__ eattr,
                                float* __restrict__ lsum, float* __restrict__ deg) {
  int e = blockIdx.x * blockDim.x + threadIdx.x;
  if (e >= NEDGES) return;
  const int d = dst[e];
  atomicAdd(&deg[d], 1.0f);
  const float* ep = eattr + (size_t)e * EDGE_DIM;
  float* lp = lsum + (size_t)d * EDGE_DIM;
#pragma unroll
  for (int c = 0; c < EDGE_DIM; ++c) atomicAdd(&lp[c], ep[c]);
}
__global__ void loop_attr_final(float* __restrict__ lsum, const float* __restrict__ deg) {
  int i = blockIdx.x * blockDim.x + threadIdx.x;
  if (i >= NNODES * EDGE_DIM) return;
  lsum[i] /= fmaxf(deg[i / EDGE_DIM], 1.0f);
}

// ---------------------------------------------------------------------------
// a_src / a_dst: per (node, head) 64-wide dot with attention vectors
// ---------------------------------------------------------------------------
__global__ void attn_node(const float* __restrict__ xl,
                          const float* __restrict__ att_src,
                          const float* __restrict__ att_dst,
                          float* __restrict__ a_src, float* __restrict__ a_dst) {
  int i = blockIdx.x * blockDim.x + threadIdx.x;
  if (i >= NNODES * HEADS) return;
  const int n = i >> 2, h = i & 3;
  const float* xp = xl + (size_t)n * WIDTH + h * HID;
  const float* as = att_src + h * HID;
  const float* ad = att_dst + h * HID;
  float s = 0.f, d = 0.f;
#pragma unroll 8
  for (int c = 0; c < HID; ++c) { float v = xp[c]; s += v * as[c]; d += v * ad[c]; }
  a_src[i] = s; a_dst[i] = d;
}

// we_att[d][h] = sum_c We[d][h*64+c] * att_edge[h][c]   (13 x 4, tiny)
__global__ void we_att_kernel(const float* __restrict__ We,
                              const float* __restrict__ att_edge,
                              float* __restrict__ weat) {
  int i = threadIdx.x;
  if (i >= EDGE_DIM * HEADS) return;
  const int d = i / HEADS, h = i % HEADS;
  float s = 0.f;
#pragma unroll 8
  for (int c = 0; c < HID; ++c) s += We[(size_t)d * WIDTH + h * HID + c] * att_edge[h * HID + c];
  weat[d * HEADS + h] = s;
}

// ---------------------------------------------------------------------------
// alpha = leaky_relu(a_src[src]+a_dst[dst]+a_edge); segment-max into keys
// Edges e<NEDGES are real; e>=NEDGES are self loops (src=dst=e-NEDGES).
// ---------------------------------------------------------------------------
__global__ void edge_alpha_max(const int* __restrict__ src, const int* __restrict__ dst,
                               const float* __restrict__ eattr,
                               const float* __restrict__ lattr,
                               const float* __restrict__ a_src,
                               const float* __restrict__ a_dst,
                               const float* __restrict__ weat,
                               float* __restrict__ alpha, unsigned* __restrict__ amaxk) {
  int e = blockIdx.x * blockDim.x + threadIdx.x;
  if (e >= NEA) return;
  int s, d; const float* ep;
  if (e < NEDGES) { s = src[e]; d = dst[e]; ep = eattr + (size_t)e * EDGE_DIM; }
  else            { s = d = e - NEDGES;     ep = lattr + (size_t)(e - NEDGES) * EDGE_DIM; }
  float ev[EDGE_DIM];
#pragma unroll
  for (int c = 0; c < EDGE_DIM; ++c) ev[c] = ep[c];
#pragma unroll
  for (int h = 0; h < HEADS; ++h) {
    float ae = 0.f;
#pragma unroll
    for (int c = 0; c < EDGE_DIM; ++c) ae += ev[c] * weat[c * HEADS + h];
    float al = a_src[s * HEADS + h] + a_dst[d * HEADS + h] + ae;
    al = (al > 0.f) ? al : NEG_SLOPE * al;
    alpha[(size_t)e * HEADS + h] = al;
    atomicMax(&amaxk[d * HEADS + h], fkey(al));
  }
}

__global__ void amax_decode(const unsigned* __restrict__ k, float* __restrict__ amax) {
  int i = blockIdx.x * blockDim.x + threadIdx.x;
  if (i >= NNODES * HEADS) return;
  float v = fkey_inv(k[i]);
  amax[i] = __builtin_isfinite(v) ? v : 0.f;   // matches reference's isfinite guard
}

// ex = exp(alpha - amax[dst]); denom[dst] += ex   (overwrite alpha with ex)
__global__ void edge_exp(const int* __restrict__ dst, float* __restrict__ alpha,
                         const float* __restrict__ amax, float* __restrict__ denom) {
  int i = blockIdx.x * blockDim.x + threadIdx.x;
  if (i >= NEA * HEADS) return;
  const int e = i >> 2, h = i & 3;
  const int d = (e < NEDGES) ? dst[e] : (e - NEDGES);
  const float ex = expf(alpha[i] - amax[d * HEADS + h]);
  alpha[i] = ex;
  atomicAdd(&denom[d * HEADS + h], ex);
}

// out[dst, h*64+c] += xl[src, h*64+c] * ex/denom[dst]
__global__ void edge_scatter(const int* __restrict__ src, const int* __restrict__ dst,
                             const float* __restrict__ xl, const float* __restrict__ ex,
                             const float* __restrict__ denom, float* __restrict__ out) {
  int i = blockIdx.x * blockDim.x + threadIdx.x;
  if (i >= NEA * HEADS) return;
  const int e = i >> 2, h = i & 3;
  const int s = (e < NEDGES) ? src[e] : (e - NEDGES);
  const int d = (e < NEDGES) ? dst[e] : (e - NEDGES);
  const float attn = ex[i] / fmaxf(denom[d * HEADS + h], 1e-16f);
  const float* xp = xl + (size_t)s * WIDTH + h * HID;
  float* op = out + (size_t)d * WIDTH + h * HID;
#pragma unroll 4
  for (int c = 0; c < HID; ++c) atomicAdd(&op[c], xp[c] * attn);
}

// ---------------------------------------------------------------------------
// h_next = relu(LayerNorm(out + bias + res) * g + b) — one 256-thread block/node
// ---------------------------------------------------------------------------
__global__ __launch_bounds__(WIDTH)
void ln_relu(const float* __restrict__ out, const float* __restrict__ bias,
             const float* __restrict__ res, const float* __restrict__ g,
             const float* __restrict__ b, float* __restrict__ hout) {
  const int n = blockIdx.x, c = threadIdx.x;
  const size_t idx = (size_t)n * WIDTH + c;
  const float v = out[idx] + bias[c] + res[idx];
  __shared__ float s1[WIDTH], s2[WIDTH];
  s1[c] = v; s2[c] = v * v;
  __syncthreads();
  for (int off = WIDTH / 2; off > 0; off >>= 1) {
    if (c < off) { s1[c] += s1[c + off]; s2[c] += s2[c + off]; }
    __syncthreads();
  }
  const float mu  = s1[0] * (1.0f / WIDTH);
  const float var = s2[0] * (1.0f / WIDTH) - mu * mu;
  const float y = (v - mu) * rsqrtf(var + 1e-5f) * g[c] + b[c];
  hout[idx] = fmaxf(y, 0.f);
}

// ---------------------------------------------------------------------------
// Graph pooling
// ---------------------------------------------------------------------------
__global__ void pool_cnt(const int* __restrict__ batch, float* __restrict__ cnt) {
  int n = blockIdx.x * blockDim.x + threadIdx.x;
  if (n < NNODES) atomicAdd(&cnt[batch[n]], 1.0f);
}
__global__ void pool_accum(const int* __restrict__ batch, const float* __restrict__ h,
                           float* __restrict__ psum, unsigned* __restrict__ pmaxk) {
  int i = blockIdx.x * blockDim.x + threadIdx.x;
  if (i >= NNODES * WIDTH) return;
  const int n = i >> 8, c = i & 255;
  const int gidx = batch[n];
  const float v = h[i];
  atomicAdd(&psum[gidx * WIDTH + c], v);
  atomicMax(&pmaxk[gidx * WIDTH + c], fkey(v));
}
__global__ void pool_final(const float* __restrict__ psum, const unsigned* __restrict__ pmaxk,
                           const float* __restrict__ cnt, const float* __restrict__ gf,
                           float* __restrict__ gbuf) {
  int i = blockIdx.x * blockDim.x + threadIdx.x;
  if (i >= NGRAPH * GDIM) return;
  const int gidx = i / GDIM, c = i % GDIM;
  float v;
  if (c < WIDTH) {
    v = psum[gidx * WIDTH + c] / fmaxf(cnt[gidx], 1.0f);
  } else if (c < 2 * WIDTH) {
    float m = fkey_inv(pmaxk[gidx * WIDTH + (c - WIDTH)]);
    v = __builtin_isfinite(m) ? m : 0.f;
  } else {
    v = gf[gidx * 14 + (c - 2 * WIDTH)];
  }
  gbuf[i] = v;
}

// ---------------------------------------------------------------------------
extern "C" void kernel_launch(void* const* d_in, const int* in_sizes, int n_in,
                              void* d_out, int out_size, void* d_ws, size_t ws_size,
                              hipStream_t stream) {
  const float* x     = (const float*)d_in[0];
  const int*   eidx  = (const int*)d_in[1];
  const float* eattr = (const float*)d_in[2];
  const int*   batch = (const int*)d_in[3];
  const float* gfeat = (const float*)d_in[4];
  const int* src = eidx;
  const int* dst = eidx + NEDGES;

  const float* fcW[4] = {(const float*)d_in[5], (const float*)d_in[7],
                         (const float*)d_in[9], (const float*)d_in[11]};
  const float* fcb[4] = {(const float*)d_in[6], (const float*)d_in[8],
                         (const float*)d_in[10], (const float*)d_in[12]};
  const int lbase[3] = {13, 23, 31};
  const float *Wl[3], *Wel[3], *adl[3], *ael[3], *asl[3], *bl[3], *lnbl[3], *lngl[3];
  for (int l = 0; l < 3; ++l) {
    Wl[l]   = (const float*)d_in[lbase[l] + 0];
    Wel[l]  = (const float*)d_in[lbase[l] + 1];
    adl[l]  = (const float*)d_in[lbase[l] + 2];
    ael[l]  = (const float*)d_in[lbase[l] + 3];
    asl[l]  = (const float*)d_in[lbase[l] + 4];
    bl[l]   = (const float*)d_in[lbase[l] + 5];
    lnbl[l] = (const float*)d_in[lbase[l] + 6];
    lngl[l] = (const float*)d_in[lbase[l] + 7];
  }
  const float* projW = (const float*)d_in[21];
  const float* projB = (const float*)d_in[22];
  const float* nodeW = (const float*)d_in[39];
  const float* nodeB = (const float*)d_in[40];

  // ---- workspace carve (~90 MB f32 + ~11 MB f16) ----
  float* w = (float*)d_ws;
  auto take = [&](size_t n) { float* p = w; w += n; return p; };
  float*    h_a   = take((size_t)NNODES * WIDTH);
  float*    h_b   = take((size_t)NNODES * WIDTH);
  float*    xl    = take((size_t)NNODES * WIDTH);
  float*    outb  = take((size_t)NNODES * WIDTH);
  float*    alpha = take((size_t)NEA * HEADS);
  float*    a_src = take((size_t)NNODES * HEADS);
  float*    a_dst = take((size_t)NNODES * HEADS);
  unsigned* amaxk = (unsigned*)take((size_t)NNODES * HEADS);
  float*    amax  = take((size_t)NNODES * HEADS);
  float*    denom = take((size_t)NNODES * HEADS);
  float*    deg   = take(NNODES);
  float*    lattr = take((size_t)NNODES * EDGE_DIM);
  float*    weat  = take(64);
  float*    cnt   = take(NGRAPH);
  float*    psum  = take((size_t)NGRAPH * WIDTH);
  unsigned* pmaxk = (unsigned*)take((size_t)NGRAPH * WIDTH);
  float*    gbuf  = take((size_t)NGRAPH * GDIM);
  float*    g1    = take((size_t)NGRAPH * 256);
  float*    g2    = take((size_t)NGRAPH * 128);
  float*    g3    = take((size_t)NGRAPH * 64);

  // f16 region (16-byte aligned: all prior sizes are multiples of 4 floats)
  _Float16* hw = (_Float16*)w;
  auto takeh = [&](size_t n) { _Float16* p = hw; hw += n; return p; };
  _Float16* Ah      = takeh((size_t)NNODES * WIDTH);   // shared activation buffer
  _Float16* btNode  = takeh(64 * 64);                  // [Np,Kp] transposed weights
  _Float16* btW0    = takeh((size_t)256 * 64);
  _Float16* btW1    = takeh((size_t)256 * 256);
  _Float16* btW2    = takeh((size_t)256 * 256);
  _Float16* btProj  = takeh((size_t)256 * 64);
  _Float16* btFc0   = takeh((size_t)256 * 544);        // Kp(526)=544
  _Float16* btFc1   = takeh((size_t)128 * 256);
  _Float16* btFc2   = takeh((size_t)64 * 128);
  _Float16* btFc3   = takeh((size_t)32 * 64);
  (void)ws_size; (void)in_sizes; (void)n_in; (void)out_size;

  const int T = 256;
  auto cvtB = [&](const float* B, _Float16* Bt, int K, int N) {
    const int Kp = padup(K, 32), Np = padup(N, 32);
    cvtBt<<<cdiv(Np * Kp, T), T, 0, stream>>>(B, Bt, K, N, Kp, Np);
  };
  auto cvtAct = [&](const float* A, int M, int K) {
    const int Kp = padup(K, 32);
    cvtA<<<cdiv(M * Kp, T), T, 0, stream>>>(A, Ah, M, K, Kp);
  };
  auto gemm = [&](const _Float16* Bt, const float* bias, float* C,
                  int M, int N, int K, int relu) {
    const int Kp = padup(K, 32), Np = padup(N, 32);
    wmma_gemm_f16<<<dim3(M / 16, Np / 32), 32, 0, stream>>>(Ah, Bt, bias, C, M, N, Kp, relu);
  };

  // ---- weight conversions (once per launch) ----
  cvtB(nodeW, btNode, NODE_DIM, HID);
  cvtB(Wl[0], btW0, HID, WIDTH);
  cvtB(Wl[1], btW1, WIDTH, WIDTH);
  cvtB(Wl[2], btW2, WIDTH, WIDTH);
  cvtB(projW, btProj, HID, WIDTH);
  cvtB(fcW[0], btFc0, GDIM, 256);
  cvtB(fcW[1], btFc1, 256, 128);
  cvtB(fcW[2], btFc2, 128, 64);
  cvtB(fcW[3], btFc3, 64, 2);
  const _Float16* btW[3] = {btW0, btW1, btW2};

  // ---- self-loop attrs ----
  fill_f32<<<cdiv(NNODES, T), T, 0, stream>>>(deg, 0.f, NNODES);
  fill_f32<<<cdiv(NNODES * EDGE_DIM, T), T, 0, stream>>>(lattr, 0.f, NNODES * EDGE_DIM);
  loop_attr_accum<<<cdiv(NEDGES, T), T, 0, stream>>>(dst, eattr, lattr, deg);
  loop_attr_final<<<cdiv(NNODES * EDGE_DIM, T), T, 0, stream>>>(lattr, deg);

  // ---- h0 = x @ node_W + node_b  [20000,58]x[58,64] ----
  cvtAct(x, NNODES, NODE_DIM);
  gemm(btNode, nodeB, h_a, NNODES, HID, NODE_DIM, 0);

  // ---- 3 GAT layers ----
  for (int l = 0; l < 3; ++l) {
    const int cin = (l == 0) ? HID : WIDTH;
    const float* hcur = (l == 1) ? h_b : h_a;   // l0:h_a  l1:h_b  l2:h_a
    float*       hnxt = (l == 1) ? h_a : h_b;   // l0:h_b  l1:h_a  l2:h_b

    cvtAct(hcur, NNODES, cin);                  // shared for xl + proj GEMMs
    gemm(btW[l], nullptr, xl, NNODES, WIDTH, cin, 0);

    const float* res;
    if (l == 0) {                               // residual projection of h0
      gemm(btProj, projB, hnxt, NNODES, WIDTH, HID, 0);
      res = hnxt;                               // ln_relu reads it in place
    } else {
      res = hcur;
    }

    attn_node<<<cdiv(NNODES * HEADS, T), T, 0, stream>>>(xl, asl[l], adl[l], a_src, a_dst);
    we_att_kernel<<<1, 64, 0, stream>>>(Wel[l], ael[l], weat);

    fill_u32<<<cdiv(NNODES * HEADS, T), T, 0, stream>>>(amaxk, 0u, NNODES * HEADS);
    edge_alpha_max<<<cdiv(NEA, T), T, 0, stream>>>(src, dst, eattr, lattr,
                                                   a_src, a_dst, weat, alpha, amaxk);
    amax_decode<<<cdiv(NNODES * HEADS, T), T, 0, stream>>>(amaxk, amax);

    fill_f32<<<cdiv(NNODES * HEADS, T), T, 0, stream>>>(denom, 0.f, NNODES * HEADS);
    edge_exp<<<cdiv(NEA * HEADS, T), T, 0, stream>>>(dst, alpha, amax, denom);

    fill_f32<<<cdiv(NNODES * WIDTH, T), T, 0, stream>>>(outb, 0.f, NNODES * WIDTH);
    edge_scatter<<<cdiv(NEA * HEADS, T), T, 0, stream>>>(src, dst, xl, alpha, denom, outb);

    ln_relu<<<NNODES, WIDTH, 0, stream>>>(outb, bl[l], res, lngl[l], lnbl[l], hnxt);
  }
  // final h is h_b

  // ---- pooling ----
  fill_f32<<<cdiv(NGRAPH, T), T, 0, stream>>>(cnt, 0.f, NGRAPH);
  fill_f32<<<cdiv(NGRAPH * WIDTH, T), T, 0, stream>>>(psum, 0.f, NGRAPH * WIDTH);
  fill_u32<<<cdiv(NGRAPH * WIDTH, T), T, 0, stream>>>(pmaxk, 0u, NGRAPH * WIDTH);
  pool_cnt<<<cdiv(NNODES, T), T, 0, stream>>>(batch, cnt);
  pool_accum<<<cdiv(NNODES * WIDTH, T), T, 0, stream>>>(batch, h_b, psum, pmaxk);
  pool_final<<<cdiv(NGRAPH * GDIM, T), T, 0, stream>>>(psum, pmaxk, cnt, gfeat, gbuf);

  // ---- FC head (WMMA, relu fused) ----
  cvtAct(gbuf, NGRAPH, GDIM);
  gemm(btFc0, fcb[0], g1, NGRAPH, 256, GDIM, 1);
  cvtAct(g1, NGRAPH, 256);
  gemm(btFc1, fcb[1], g2, NGRAPH, 128, 256, 1);
  cvtAct(g2, NGRAPH, 128);
  gemm(btFc2, fcb[2], g3, NGRAPH, 64, 128, 1);
  cvtAct(g3, NGRAPH, 64);
  gemm(btFc3, fcb[3], (float*)d_out, NGRAPH, 2, 64, 0);
}
